// GAT_block_5428838662666
// MI455X (gfx1250) — matrix operand order repaired and checked by reading, not compiled, not run on hardware
//
#include <hip/hip_runtime.h>
#include <hip/hip_bf16.h>
#include <math.h>

// ---------------------------------------------------------------------------
// Transformer block (rmsnorm -> QKV -> masked attention -> add&norm ->
// SwiGLU-ish FFN -> residual) for B=2, N=2048, D=512, H_DIM=2048, HEADS=8.
// All GEMM math on v_wmma_f32_16x16x32_f16 (f16 in, f32 accumulate).
// Mask pre-packed int32 -> transposed u8 [key][qrow] so each lane's 8 query
// rows are one aligned 8-byte LDS read; tiles staged through LDS;
// global_prefetch_b8 hints for the next tile.
// ---------------------------------------------------------------------------

typedef __attribute__((ext_vector_type(16))) _Float16 v16h;
typedef __attribute__((ext_vector_type(8)))  _Float16 v8h;
typedef __attribute__((ext_vector_type(8)))  float    v8f;

#define SEQ_N 2048
#define DIM_D 512
#define HID_H 2048

__device__ __forceinline__ int lane_id() { return threadIdx.x & 31; }

// ---- WMMA fragment loaders (contiguous k = 16*(L/16)+s, same for A and B) -----

// A matrix 16x32 f16: lane L holds row M = L&15.
// lanes 0-15: elements 0..15 -> K=0..15
// lanes16-31: elements 0..15 -> K=16..31
__device__ __forceinline__ v16h load_a_frag(const _Float16* base, int row0,
                                            int k0, int lds) {
  const int lane = lane_id();
  const int m = lane & 15;
  const int kb = (lane >> 4) << 4;  // 0 or 16
  const _Float16* p = base + (size_t)(row0 + m) * lds + k0 + kb;
  v8h lo = *(const v8h*)(p);
  v8h hi = *(const v8h*)(p + 8);
  v16h r;
#pragma unroll
  for (int i = 0; i < 8; ++i) { r[i] = lo[i]; r[i + 8] = hi[i]; }
  return r;
}

// B matrix 32x16 f16, with B[k][n] stored as rows of Wrow[n][k] (torch Linear
// weight layout). lane L holds column N = L&15; lanes 0-15 hold K=0..15,
// lanes 16-31 hold K=16..31.
__device__ __forceinline__ v16h load_b_frag(const _Float16* base, int col0,
                                            int k0, int lds) {
  const int lane = lane_id();
  const int n = lane & 15;
  const int kb = (lane >> 4) << 4;  // 0 or 16
  const _Float16* p = base + (size_t)(col0 + n) * lds + k0 + kb;
  v8h lo = *(const v8h*)(p);
  v8h hi = *(const v8h*)(p + 8);
  v16h r;
#pragma unroll
  for (int i = 0; i < 8; ++i) { r[i] = lo[i]; r[i + 8] = hi[i]; }
  return r;
}

// Same B fragment but sourced from a [k][n]-major LDS tile (V matrix).
__device__ __forceinline__ v16h load_bT_frag(const _Float16* base, int k0,
                                             int col0, int lds) {
  const int lane = lane_id();
  const int n = lane & 15;
  const int kb = (lane >> 4) << 4;
  const _Float16* p = base + (size_t)(k0 + kb) * lds + col0 + n;
  v16h r;
#pragma unroll
  for (int i = 0; i < 16; ++i) r[i] = p[i * lds];
  return r;
}

__device__ __forceinline__ v8f wmma_f16(v16h a, v16h b, v8f c) {
  return __builtin_amdgcn_wmma_f32_16x16x32_f16(false, a, false, b, (short)0,
                                                c, false, false);
}

// ---------------------------------------------------------------------------
// Generic tiled GEMM: out[m,n] = epilogue( (sum_k A[m,k]*W[n,k] + bias)*scale )
// Block tile 128x64, 8 waves (4x2), wave tile 32x32 (4 WMMA accumulators),
// K staged through LDS 64 at a time.
// EPI 0: out_h = (acc + bias0) * scale              (QKV; q gets scale=0.125)
// EPI 1: out_h = silu(acc0+bias0) * (acc1+bias1)    (fused FFN gate, dual B)
// EPI 2: out_f = res + acc + bias0                  (final projection)
// ---------------------------------------------------------------------------
template <int EPI>
__global__ __launch_bounds__(256) void gemm_f16_kernel(
    const _Float16* __restrict__ A, const _Float16* __restrict__ W0,
    const _Float16* __restrict__ W1, const float* __restrict__ bias0,
    const float* __restrict__ bias1, const float* __restrict__ res,
    _Float16* __restrict__ outh, float* __restrict__ outf, int M, int N,
    int K, float scale) {
  __shared__ __align__(16) _Float16 As[128][72];
  __shared__ __align__(16) _Float16 Bs0[64][72];
  __shared__ __align__(16) _Float16 Bs1[64][72];

  const int tid = threadIdx.x;
  const int wid = tid >> 5;
  const int wm = wid >> 1;  // 0..3
  const int wn = wid & 1;   // 0..1
  const int mbase = blockIdx.y * 128;
  const int nbase = blockIdx.x * 64;

  const v8f vzero = {0.f, 0.f, 0.f, 0.f, 0.f, 0.f, 0.f, 0.f};
  v8f acc0[4] = {vzero, vzero, vzero, vzero};
  v8f acc1[4] = {vzero, vzero, vzero, vzero};

  for (int k0 = 0; k0 < K; k0 += 64) {
    __syncthreads();
#pragma unroll
    for (int i = 0; i < 4; ++i) {  // A tile: 128 rows x 64 halves
      int c = tid + i * 256;
      int row = c >> 3, col8 = (c & 7) << 3;
      *(uint4*)&As[row][col8] =
          *(const uint4*)(A + (size_t)(mbase + row) * K + k0 + col8);
    }
#pragma unroll
    for (int i = 0; i < 2; ++i) {  // B tile(s): 64 rows x 64 halves
      int c = tid + i * 256;
      int row = c >> 3, col8 = (c & 7) << 3;
      *(uint4*)&Bs0[row][col8] =
          *(const uint4*)(W0 + (size_t)(nbase + row) * K + k0 + col8);
      if (EPI == 1)
        *(uint4*)&Bs1[row][col8] =
            *(const uint4*)(W1 + (size_t)(nbase + row) * K + k0 + col8);
    }
    // prefetch next K-slice (global_prefetch_b8)
    if (k0 + 64 < K) {
      int row = tid >> 3, col8 = (tid & 7) << 3;
      __builtin_prefetch(A + (size_t)(mbase + row) * K + k0 + 64 + col8, 0, 3);
      if (row < 64)
        __builtin_prefetch(W0 + (size_t)(nbase + row) * K + k0 + 64 + col8, 0,
                           3);
    }
    __syncthreads();

#pragma unroll
    for (int ks = 0; ks < 64; ks += 32) {
      v16h a0 = load_a_frag(&As[0][0], wm * 32, ks, 72);
      v16h a1 = load_a_frag(&As[0][0], wm * 32 + 16, ks, 72);
      v16h b0 = load_b_frag(&Bs0[0][0], wn * 32, ks, 72);
      v16h b1 = load_b_frag(&Bs0[0][0], wn * 32 + 16, ks, 72);
      acc0[0] = wmma_f16(a0, b0, acc0[0]);
      acc0[1] = wmma_f16(a0, b1, acc0[1]);
      acc0[2] = wmma_f16(a1, b0, acc0[2]);
      acc0[3] = wmma_f16(a1, b1, acc0[3]);
      if (EPI == 1) {
        v16h c0 = load_b_frag(&Bs1[0][0], wn * 32, ks, 72);
        v16h c1 = load_b_frag(&Bs1[0][0], wn * 32 + 16, ks, 72);
        acc1[0] = wmma_f16(a0, c0, acc1[0]);
        acc1[1] = wmma_f16(a0, c1, acc1[1]);
        acc1[2] = wmma_f16(a1, c0, acc1[2]);
        acc1[3] = wmma_f16(a1, c1, acc1[3]);
      }
    }
  }

  const int lane = lane_id();
  const int nloc = lane & 15;
  const int rhalf = (lane >> 4) << 3;  // +8 rows in upper half-wave
#pragma unroll
  for (int f = 0; f < 4; ++f) {
    int frow0 = mbase + wm * 32 + (f >> 1) * 16 + rhalf;
    int col = nbase + wn * 32 + (f & 1) * 16 + nloc;
#pragma unroll
    for (int r = 0; r < 8; ++r) {
      int row = frow0 + r;
      float v = acc0[f][r] + bias0[col];
      if (EPI == 0) {
        outh[(size_t)row * N + col] = (_Float16)(v * scale);
      } else if (EPI == 1) {
        float v2 = acc1[f][r] + bias1[col];
        float s = v / (1.f + __expf(-v));  // silu
        outh[(size_t)row * N + col] = (_Float16)(s * v2);
      } else {
        outf[(size_t)row * N + col] = res[(size_t)row * N + col] + v;
      }
    }
  }
}

// ---------------------------------------------------------------------------
// Flash attention: one block per (b, head, 64-query tile); 128 threads =
// 4 waves, each wave owns 16 query rows x full 64-key tile (4 S fragments)
// and keeps a 16x64 f32 output accumulator across the key loop. Mask is
// byte-packed AND transposed ([key][qrow]) so each lane reads its 8 query
// rows as one aligned uint2 from LDS. Q is pre-scaled by 1/8.
// ---------------------------------------------------------------------------
__global__ __launch_bounds__(128) void attn_kernel(
    const _Float16* __restrict__ Q, const _Float16* __restrict__ Km,
    const _Float16* __restrict__ Vm, const unsigned char* __restrict__ MpkT,
    float* __restrict__ O) {
  __shared__ __align__(16) _Float16 Qs[64][72];
  __shared__ __align__(16) _Float16 Ks[64][72];
  __shared__ __align__(16) _Float16 Vs[64][72];
  __shared__ __align__(16) _Float16 Ps[4][16][72];
  __shared__ __align__(16) unsigned char MsT[64][64];  // [key][qrow]

  const int tid = threadIdx.x;
  const int w = tid >> 5;
  const int lane = tid & 31;
  const int qbase = blockIdx.x * 64;
  const int head = blockIdx.y;
  const int b = blockIdx.z;
  const size_t rowbase = (size_t)b * SEQ_N;
  const int hcol = head * 64;

#pragma unroll
  for (int i = 0; i < 4; ++i) {  // load Q tile 64x64
    int c = tid + i * 128;
    int row = c >> 3, col8 = (c & 7) << 3;
    *(uint4*)&Qs[row][col8] =
        *(const uint4*)(Q + (rowbase + qbase + row) * DIM_D + hcol + col8);
  }
  __syncthreads();

  v16h qa0 = load_a_frag(&Qs[0][0], w * 16, 0, 72);
  v16h qa1 = load_a_frag(&Qs[0][0], w * 16, 32, 72);

  const v8f vzero = {0.f, 0.f, 0.f, 0.f, 0.f, 0.f, 0.f, 0.f};
  v8f oacc[4] = {vzero, vzero, vzero, vzero};
  float mrow[8], lrow[8];
#pragma unroll
  for (int r = 0; r < 8; ++r) { mrow[r] = -3.0e38f; lrow[r] = 0.f; }

  const int mqoff = w * 16 + ((lane >> 4) << 3);  // this lane's 8 query rows

  for (int kt = 0; kt < SEQ_N / 64; ++kt) {
    __syncthreads();  // previous-iteration readers done
#pragma unroll
    for (int i = 0; i < 4; ++i) {  // load K,V tiles 64x64
      int c = tid + i * 128;
      int row = c >> 3, col8 = (c & 7) << 3;
      *(uint4*)&Ks[row][col8] = *(const uint4*)(
          Km + (rowbase + kt * 64 + row) * DIM_D + hcol + col8);
      *(uint4*)&Vs[row][col8] = *(const uint4*)(
          Vm + (rowbase + kt * 64 + row) * DIM_D + hcol + col8);
    }
#pragma unroll
    for (int i = 0; i < 2; ++i) {  // load transposed mask tile (64x64 bytes)
      int c = tid + i * 128;
      int row = c >> 2, col16 = (c & 3) << 4;
      *(uint4*)&MsT[row][col16] = *(const uint4*)(
          MpkT + (rowbase + kt * 64 + row) * SEQ_N + qbase + col16);
    }
    if (kt + 1 < SEQ_N / 64) {  // prefetch next tiles
      int row = tid >> 3, col8 = (tid & 7) << 3;
      __builtin_prefetch(
          Km + (rowbase + (kt + 1) * 64 + row) * DIM_D + hcol + col8, 0, 3);
      __builtin_prefetch(
          Vm + (rowbase + (kt + 1) * 64 + row) * DIM_D + hcol + col8, 0, 3);
      __builtin_prefetch(
          MpkT + (rowbase + (kt + 1) * 64 + (tid >> 1)) * SEQ_N + qbase, 0, 3);
    }
    __syncthreads();

    // S = (Q/8) K^T : four independent WMMA chains first...
    v8f sacc[4];
#pragma unroll
    for (int j = 0; j < 4; ++j) {
      v16h kb0 = load_b_frag(&Ks[0][0], j * 16, 0, 72);
      v16h kb1 = load_b_frag(&Ks[0][0], j * 16, 32, 72);
      v8f s = wmma_f16(qa0, kb0, vzero);
      sacc[j] = wmma_f16(qa1, kb1, s);
    }
    // ...then apply the mask (one uint2 LDS read per fragment per lane)
    float sfr[4][8];
#pragma unroll
    for (int j = 0; j < 4; ++j) {
      uint2 mb = *(const uint2*)&MsT[j * 16 + (lane & 15)][mqoff];
#pragma unroll
      for (int r = 0; r < 8; ++r) {
        unsigned int byte =
            ((r < 4 ? (mb.x >> (8 * r)) : (mb.y >> (8 * (r - 4)))) & 0xffu);
        sfr[j][r] = byte ? sacc[j][r] : -1e20f;
      }
    }

    // online softmax: row stats live per-lane for 8 rows; xor-shuffles over
    // the 16-lane half-wave reduce across the key dimension.
    float alpha[8];
#pragma unroll
    for (int r = 0; r < 8; ++r) {
      float tmax = fmaxf(fmaxf(sfr[0][r], sfr[1][r]),
                         fmaxf(sfr[2][r], sfr[3][r]));
#pragma unroll
      for (int off = 1; off < 16; off <<= 1)
        tmax = fmaxf(tmax, __shfl_xor(tmax, off, 32));
      float mnew = fmaxf(mrow[r], tmax);
      alpha[r] = __expf(mrow[r] - mnew);
      mrow[r] = mnew;
      float tsum = 0.f;
#pragma unroll
      for (int j = 0; j < 4; ++j) {
        float p = __expf(sfr[j][r] - mnew);
        sfr[j][r] = p;
        tsum += p;
      }
#pragma unroll
      for (int off = 1; off < 16; off <<= 1)
        tsum += __shfl_xor(tsum, off, 32);
      lrow[r] = lrow[r] * alpha[r] + tsum;
    }

    // P (C-layout) -> LDS -> A-layout fragments
#pragma unroll
    for (int j = 0; j < 4; ++j)
#pragma unroll
      for (int r = 0; r < 8; ++r)
        Ps[w][r + ((lane >> 4) << 3)][j * 16 + (lane & 15)] =
            (_Float16)sfr[j][r];
    __syncthreads();

#pragma unroll
    for (int j = 0; j < 4; ++j)
#pragma unroll
      for (int r = 0; r < 8; ++r) oacc[j][r] *= alpha[r];

    v16h pa0 = load_a_frag(&Ps[w][0][0], 0, 0, 72);
    v16h pa1 = load_a_frag(&Ps[w][0][0], 0, 32, 72);
#pragma unroll
    for (int j = 0; j < 4; ++j) {
      v16h vb0 = load_bT_frag(&Vs[0][0], 0, j * 16, 72);
      v16h vb1 = load_bT_frag(&Vs[0][0], 32, j * 16, 72);
      oacc[j] = wmma_f16(pa0, vb0, oacc[j]);
      oacc[j] = wmma_f16(pa1, vb1, oacc[j]);
    }
  }

#pragma unroll
  for (int j = 0; j < 4; ++j) {
#pragma unroll
    for (int r = 0; r < 8; ++r) {
      int row = qbase + w * 16 + r + ((lane >> 4) << 3);
      int col = hcol + j * 16 + (lane & 15);
      O[(rowbase + row) * DIM_D + col] = oacc[j][r] / lrow[r];
    }
  }
}

// ---------------------------------------------------------------------------
// rmsnorm (one block per row, 256 threads x 2 elements) and add+norm
// ---------------------------------------------------------------------------
__global__ __launch_bounds__(256) void rmsnorm_kernel(
    const float* __restrict__ x, const float* __restrict__ g,
    _Float16* __restrict__ out) {
  __shared__ float red[256];
  const int row = blockIdx.x;
  const int tid = threadIdx.x;
  const size_t base = (size_t)row * DIM_D;
  float v0 = x[base + tid], v1 = x[base + tid + 256];
  red[tid] = v0 * v0 + v1 * v1;
  __syncthreads();
  for (int s = 128; s > 0; s >>= 1) {
    if (tid < s) red[tid] += red[tid + s];
    __syncthreads();
  }
  float rs = rsqrtf(red[0] * (1.f / DIM_D) + 1.1920929e-07f);
  out[base + tid] = (_Float16)(v0 * rs * g[tid]);
  out[base + tid + 256] = (_Float16)(v1 * rs * g[tid + 256]);
}

__global__ __launch_bounds__(256) void addnorm_kernel(
    const float* __restrict__ x, const float* __restrict__ o,
    const float* __restrict__ g, float* __restrict__ h,
    _Float16* __restrict__ hn) {
  __shared__ float red[256];
  const int row = blockIdx.x;
  const int tid = threadIdx.x;
  const size_t base = (size_t)row * DIM_D;
  float v0 = x[base + tid] + o[base + tid];
  float v1 = x[base + tid + 256] + o[base + tid + 256];
  h[base + tid] = v0;
  h[base + tid + 256] = v1;
  red[tid] = v0 * v0 + v1 * v1;
  __syncthreads();
  for (int s = 128; s > 0; s >>= 1) {
    if (tid < s) red[tid] += red[tid + s];
    __syncthreads();
  }
  float rs = rsqrtf(red[0] * (1.f / DIM_D) + 1.1920929e-07f);
  hn[base + tid] = (_Float16)(v0 * rs * g[tid]);
  hn[base + tid + 256] = (_Float16)(v1 * rs * g[tid + 256]);
}

__global__ void cvt_f16_kernel(const float* __restrict__ in,
                               _Float16* __restrict__ out, int n) {
  int i = blockIdx.x * 256 + threadIdx.x;
  if (i < n) out[i] = (_Float16)in[i];
}

// pack int32 0/1 mask -> bytes, transposed per batch: out[b][key][qrow]
__global__ __launch_bounds__(256) void pack_mask_T_kernel(
    const int* __restrict__ DA, unsigned char* __restrict__ outT) {
  __shared__ __align__(16) unsigned char Mt[64][64];
  const int q0 = blockIdx.x * 64;
  const int k0 = blockIdx.y * 64;
  const int b = blockIdx.z;
  const int tid = threadIdx.x;
  const size_t base = (size_t)b * SEQ_N * SEQ_N;
#pragma unroll
  for (int i = 0; i < 4; ++i) {  // 64x64 ints, int4-coalesced
    int c = tid + i * 256;
    int row = c >> 4, col4 = (c & 15) << 2;
    int4 v =
        *(const int4*)(DA + base + (size_t)(q0 + row) * SEQ_N + k0 + col4);
    uchar4 u;
    u.x = (unsigned char)v.x;
    u.y = (unsigned char)v.y;
    u.z = (unsigned char)v.z;
    u.w = (unsigned char)v.w;
    *(uchar4*)&Mt[row][col4] = u;
  }
  __syncthreads();
  {  // write transposed, uint4-coalesced along qrow
    int key = tid >> 2, q16 = (tid & 3) << 4;
    alignas(16) unsigned char tmp[16];
#pragma unroll
    for (int ii = 0; ii < 16; ++ii) tmp[ii] = Mt[q16 + ii][key];
    *(uint4*)(outT + base + (size_t)(k0 + key) * SEQ_N + q0 + q16) =
        *(const uint4*)tmp;
  }
}

// ---------------------------------------------------------------------------
extern "C" void kernel_launch(void* const* d_in, const int* in_sizes, int n_in,
                              void* d_out, int out_size, void* d_ws,
                              size_t ws_size, hipStream_t stream) {
  const float* x  = (const float*)d_in[0];
  const int*   DA = (const int*)d_in[1];
  const float* Wq = (const float*)d_in[2];
  const float* bq = (const float*)d_in[3];
  const float* Wk = (const float*)d_in[4];
  const float* bk = (const float*)d_in[5];
  const float* Wv = (const float*)d_in[6];
  const float* bv = (const float*)d_in[7];
  const float* W1 = (const float*)d_in[8];
  const float* b1 = (const float*)d_in[9];
  const float* W2 = (const float*)d_in[10];
  const float* b2 = (const float*)d_in[11];
  const float* W3 = (const float*)d_in[12];
  const float* b3 = (const float*)d_in[13];
  const float* g1 = (const float*)d_in[14];
  const float* g2 = (const float*)d_in[15];
  float* out = (float*)d_out;

  const int ROWS = 2 * SEQ_N;  // 4096

  char* ws = (char*)d_ws;
  size_t off = 0;
  auto allocH = [&](size_t n) {
    _Float16* p = (_Float16*)(ws + off);
    off += n * sizeof(_Float16);
    return p;
  };
  auto allocF = [&](size_t n) {
    float* p = (float*)(ws + off);
    off += n * sizeof(float);
    return p;
  };

  _Float16* Wq16 = allocH((size_t)DIM_D * DIM_D);
  _Float16* Wk16 = allocH((size_t)DIM_D * DIM_D);
  _Float16* Wv16 = allocH((size_t)DIM_D * DIM_D);
  _Float16* W116 = allocH((size_t)HID_H * DIM_D);
  _Float16* W216 = allocH((size_t)HID_H * DIM_D);
  _Float16* W316 = allocH((size_t)DIM_D * HID_H);
  _Float16* xn16 = allocH((size_t)ROWS * DIM_D);
  _Float16* q16  = allocH((size_t)ROWS * DIM_D);
  _Float16* k16  = allocH((size_t)ROWS * DIM_D);
  _Float16* v16  = allocH((size_t)ROWS * DIM_D);
  _Float16* hn16 = allocH((size_t)ROWS * DIM_D);
  _Float16* gate16 = allocH((size_t)ROWS * HID_H);
  float* o32 = allocF((size_t)ROWS * DIM_D);
  float* h32 = allocF((size_t)ROWS * DIM_D);
  unsigned char* mpkT = (unsigned char*)(ws + off);
  off += (size_t)2 * SEQ_N * SEQ_N;  // packed transposed mask bytes
  (void)ws_size;

  // 0) pack + transpose mask int32 -> u8 [b][key][qrow]
  pack_mask_T_kernel<<<dim3(SEQ_N / 64, SEQ_N / 64, 2), 256, 0, stream>>>(
      DA, mpkT);

  // 1) weights -> f16
  const int nDD = DIM_D * DIM_D, nHD = HID_H * DIM_D;
  cvt_f16_kernel<<<nDD / 256, 256, 0, stream>>>(Wq, Wq16, nDD);
  cvt_f16_kernel<<<nDD / 256, 256, 0, stream>>>(Wk, Wk16, nDD);
  cvt_f16_kernel<<<nDD / 256, 256, 0, stream>>>(Wv, Wv16, nDD);
  cvt_f16_kernel<<<nHD / 256, 256, 0, stream>>>(W1, W116, nHD);
  cvt_f16_kernel<<<nHD / 256, 256, 0, stream>>>(W2, W216, nHD);
  cvt_f16_kernel<<<nHD / 256, 256, 0, stream>>>(W3, W316, nHD);

  // 2) xn = rmsnorm(x, g1)
  rmsnorm_kernel<<<ROWS, 256, 0, stream>>>(x, g1, xn16);

  // 3) QKV projections (q pre-scaled by 1/sqrt(head_dim) = 0.125)
  dim3 gqkv(DIM_D / 64, ROWS / 128);
  gemm_f16_kernel<0><<<gqkv, 256, 0, stream>>>(xn16, Wq16, nullptr, bq,
                                               nullptr, nullptr, q16, nullptr,
                                               ROWS, DIM_D, DIM_D, 0.125f);
  gemm_f16_kernel<0><<<gqkv, 256, 0, stream>>>(xn16, Wk16, nullptr, bk,
                                               nullptr, nullptr, k16, nullptr,
                                               ROWS, DIM_D, DIM_D, 1.0f);
  gemm_f16_kernel<0><<<gqkv, 256, 0, stream>>>(xn16, Wv16, nullptr, bv,
                                               nullptr, nullptr, v16, nullptr,
                                               ROWS, DIM_D, DIM_D, 1.0f);

  // 4) masked flash attention -> o32
  attn_kernel<<<dim3(SEQ_N / 64, 8, 2), 128, 0, stream>>>(q16, k16, v16,
                                                          mpkT, o32);

  // 5) h = x + o ; hn = rmsnorm(h, g2)
  addnorm_kernel<<<ROWS, 256, 0, stream>>>(x, o32, g2, h32, hn16);

  // 6) gate = silu(hn W1^T + b1) * (hn W2^T + b2)   (dual-B fused GEMM)
  gemm_f16_kernel<1><<<dim3(HID_H / 64, ROWS / 128), 256, 0, stream>>>(
      hn16, W116, W216, b1, b2, nullptr, gate16, nullptr, ROWS, HID_H, DIM_D,
      1.0f);

  // 7) out = h + gate W3^T + b3
  gemm_f16_kernel<2><<<dim3(DIM_D / 64, ROWS / 128), 256, 0, stream>>>(
      gate16, W316, nullptr, b3, nullptr, h32, nullptr, out, ROWS, DIM_D,
      HID_H, 1.0f);
  (void)in_sizes; (void)n_in; (void)out_size;
}
